// MoEBlock_64080912056798
// MI455X (gfx1250) — matrix operand order repaired and checked by reading, not compiled, not run on hardware
//
#include <hip/hip_runtime.h>

// ---------------- problem constants ----------------
constexpr int BTOK = 8192;   // tokens
constexpr int DDIM = 1024;   // model dim
constexpr int HDIM = 256;    // hidden dim
constexpr int NEXP = 16;     // experts
constexpr int TOPK = 4;
constexpr int BM   = 64;                 // tokens per expert tile
constexpr int TILES_PER_E = BTOK / BM;   // worst case all tokens -> one expert

// ---------------- WMMA types ----------------
typedef __attribute__((ext_vector_type(16))) __bf16 v16bf;
typedef __attribute__((ext_vector_type(8)))  float  v8f;

union FragU { v16bf v; uint4 q[2]; };

__device__ __forceinline__ unsigned short f2bf(float f) {
  unsigned u = __float_as_uint(f);
  unsigned r = (u + 0x7FFFu + ((u >> 16) & 1u)) >> 16;
  return (unsigned short)r;
}

// ---- CDNA5 async global->LDS copy (ASYNCcnt-tracked, §15.18.3 op 98) ----
__device__ __forceinline__ void async_copy_b128(const void* gsrc, void* lds_dst) {
  unsigned lds = (unsigned)(uintptr_t)lds_dst;              // low 32 bits = LDS byte addr
  unsigned long long ga = (unsigned long long)(uintptr_t)gsrc;
  asm volatile("global_load_async_to_lds_b128 %0, %1, off"
               :: "v"(lds), "v"(ga) : "memory");
}
__device__ __forceinline__ void wait_async0() {
  asm volatile("s_wait_asynccnt 0x0" ::: "memory");
}

// ---------------- init: zero out + expert counters ----------------
__global__ void moe_init(float* __restrict__ out, int* __restrict__ ecount, long n) {
  long i = (long)blockIdx.x * blockDim.x + threadIdx.x;
  long stride = (long)gridDim.x * blockDim.x;
  for (; i < n; i += stride) out[i] = 0.0f;
  if (blockIdx.x == 0 && threadIdx.x < NEXP) ecount[threadIdx.x] = 0;
}

// ---------------- f32 -> bf16 conversion ----------------
__global__ void moe_cvt(const float* __restrict__ src, unsigned short* __restrict__ dst, long n) {
  long i = (long)blockIdx.x * blockDim.x + threadIdx.x;
  long stride = (long)gridDim.x * blockDim.x;
  for (; i < n; i += stride) dst[i] = f2bf(src[i]);
}

// ---------------- router: logits -> top4 softmax -> expert lists ----------------
__global__ __launch_bounds__(256)
void moe_router(const float* __restrict__ x, const float* __restrict__ rw,
                int* __restrict__ ecount, int* __restrict__ etok, float* __restrict__ ewt) {
  const int lane = threadIdx.x & 31;
  const int wave = threadIdx.x >> 5;
  const int tok  = blockIdx.x * 8 + wave;
  if (tok >= BTOK) return;

  float acc[NEXP];
#pragma unroll
  for (int e = 0; e < NEXP; e++) acc[e] = 0.0f;

  const float* xr = x + (long)tok * DDIM;
  for (int d = lane; d < DDIM; d += 32) {
    float xv = xr[d];
#pragma unroll
    for (int e = 0; e < NEXP; e++) acc[e] += xv * rw[e * DDIM + d];
  }
#pragma unroll
  for (int e = 0; e < NEXP; e++) {
    float v = acc[e];
    v += __shfl_down(v, 16, 32);
    v += __shfl_down(v,  8, 32);
    v += __shfl_down(v,  4, 32);
    v += __shfl_down(v,  2, 32);
    v += __shfl_down(v,  1, 32);
    acc[e] = v;
  }
  if (lane == 0) {
    int   idx[TOPK];
    float val[TOPK];
    bool  used[NEXP];
#pragma unroll
    for (int j = 0; j < NEXP; j++) used[j] = false;
    for (int k = 0; k < TOPK; k++) {
      int bi = 0; float bv = -3.4e38f;
      for (int j = 0; j < NEXP; j++) {
        if (!used[j] && acc[j] > bv) { bv = acc[j]; bi = j; }
      }
      used[bi] = true; idx[k] = bi; val[k] = bv;
    }
    float mx = val[0];           // largest picked first
    float w[TOPK], s = 0.0f;
#pragma unroll
    for (int k = 0; k < TOPK; k++) { w[k] = __expf(val[k] - mx); s += w[k]; }
    float inv = 1.0f / s;
#pragma unroll
    for (int k = 0; k < TOPK; k++) {
      int e = idx[k];
      int pos = atomicAdd(&ecount[e], 1);
      etok[(long)e * BTOK + pos] = tok;
      ewt [(long)e * BTOK + pos] = w[k] * inv;
    }
  }
}

// ---------------- fused expert FFN: relu(relu(X W1^T + b1) W2^T + b2) * wt ----------------
__global__ __launch_bounds__(256)
void moe_expert(const unsigned short* __restrict__ xb,
                const unsigned short* __restrict__ w1b,
                const float* __restrict__ b1,
                const unsigned short* __restrict__ w2b,
                const float* __restrict__ b2,
                const int* __restrict__ ecount,
                const int* __restrict__ etok,
                const float* __restrict__ ewt,
                float* __restrict__ out) {
  __shared__ alignas(16) unsigned short sA[BM * 64];      // 8 KB  X tile (64 x 64)
  __shared__ alignas(16) unsigned short sB[256 * 64];     // 32 KB W tile (256 x 64)
  __shared__ alignas(16) unsigned short sH[BM * HDIM];    // 32 KB H tile (64 x 256) bf16
  __shared__ int   sTok[BM];
  __shared__ float sWt[BM];

  const int e     = blockIdx.y;
  const int count = ecount[e];
  const int start = blockIdx.x * BM;
  if (start >= count) return;

  const int tid  = threadIdx.x;
  const int lane = tid & 31;
  const int wave = tid >> 5;
  const int hh   = lane >> 4;       // half-wave (K-group select)
  const int ln16 = lane & 15;
  const int mt   = wave & 3;        // M-tile 0..3  (rows mt*16 .. +16)
  const int ntb  = (wave >> 2) * 8; // N-tile base 0 or 8

  if (tid < BM) {
    int s  = start + tid;
    int cs = (s < count) ? s : (count - 1);   // pad with duplicate row, weight 0
    sTok[tid] = etok[(long)e * BTOK + cs];
    sWt [tid] = (s < count) ? ewt[(long)e * BTOK + s] : 0.0f;
  }
  __syncthreads();

  const v8f vzero = {0.f, 0.f, 0.f, 0.f, 0.f, 0.f, 0.f, 0.f};
  v8f c[8];
#pragma unroll
  for (int i = 0; i < 8; i++) c[i] = vzero;

  // ===== Phase 1: H(64x256) = relu(X(64x1024) @ W1e^T + b1) =====
  const unsigned short* w1e = w1b + (size_t)e * HDIM * DDIM;
  for (int kc = 0; kc < DDIM; kc += 64) {
    __syncthreads();
    // stage W1 tile via CDNA5 async global->LDS: 256 rows x 64 cols = 2048 x b128
    for (int idx = tid; idx < (256 * 64) / 8; idx += 256) {
      int n  = idx >> 3;
      int c8 = (idx & 7) * 8;
      async_copy_b128(w1e + (size_t)n * DDIM + kc + c8, &sB[n * 64 + c8]);
    }
    // stage X tile (indirect row gather): 64 rows x 64 cols = 512 x b128
    for (int idx = tid; idx < (BM * 64) / 8; idx += 256) {
      int row = idx >> 3;
      int c8  = (idx & 7) * 8;
      *(uint4*)(&sA[row * 64 + c8]) =
          *(const uint4*)(xb + (size_t)sTok[row] * DDIM + kc + c8);
    }
    wait_async0();
    __syncthreads();
#pragma unroll
    for (int kk = 0; kk < 64; kk += 32) {
      FragU a;
      const unsigned short* ab = &sA[(mt * 16 + ln16) * 64 + kk];
      a.q[0] = *(const uint4*)(ab + hh * 8);        // VGPR0-3: K = half*8 .. +8
      a.q[1] = *(const uint4*)(ab + 16 + hh * 8);   // VGPR4-7: K = 16+half*8 .. +8
      FragU bf[8];
#pragma unroll
      for (int i = 0; i < 8; i++) {                 // batch the ds loads...
        const unsigned short* bb = &sB[((ntb + i) * 16 + ln16) * 64 + kk];
        bf[i].q[0] = *(const uint4*)(bb + hh * 8);
        bf[i].q[1] = *(const uint4*)(bb + 16 + hh * 8);
      }
#pragma unroll
      for (int i = 0; i < 8; i++)                   // ...then run WMMAs back-to-back
        c[i] = __builtin_amdgcn_wmma_f32_16x16x32_bf16(
            false, a.v, false, bf[i].v, (short)0, c[i], false, false);
    }
  }
  // epilogue 1: bias + relu, write bf16 H to LDS
#pragma unroll
  for (int i = 0; i < 8; i++) {
    int n = (ntb + i) * 16 + ln16;
    float bias = b1[e * HDIM + n];
#pragma unroll
    for (int v = 0; v < 8; v++) {
      int m = mt * 16 + hh * 8 + v;       // C layout: M = half*8 + vgpr
      float hv = c[i][v] + bias;
      hv = hv > 0.0f ? hv : 0.0f;
      sH[m * HDIM + n] = f2bf(hv);
    }
  }
  __syncthreads();

  // ===== Phase 2: Y(64x1024) = relu(H(64x256) @ W2e^T + b2) * wt, scatter-add =====
  const unsigned short* w2e = w2b + (size_t)e * DDIM * HDIM;
  for (int nchunk = 0; nchunk < DDIM; nchunk += 256) {
#pragma unroll
    for (int i = 0; i < 8; i++) c[i] = vzero;
    for (int kc = 0; kc < HDIM; kc += 64) {
      __syncthreads();
      for (int idx = tid; idx < (256 * 64) / 8; idx += 256) {
        int n  = idx >> 3;
        int c8 = (idx & 7) * 8;
        async_copy_b128(w2e + (size_t)(nchunk + n) * HDIM + kc + c8, &sB[n * 64 + c8]);
      }
      wait_async0();
      __syncthreads();
#pragma unroll
      for (int kk = 0; kk < 64; kk += 32) {
        FragU a;
        const unsigned short* ab = &sH[(mt * 16 + ln16) * HDIM + kc + kk];
        a.q[0] = *(const uint4*)(ab + hh * 8);
        a.q[1] = *(const uint4*)(ab + 16 + hh * 8);
        FragU bf[8];
#pragma unroll
        for (int i = 0; i < 8; i++) {
          const unsigned short* bb = &sB[((ntb + i) * 16 + ln16) * 64 + kk];
          bf[i].q[0] = *(const uint4*)(bb + hh * 8);
          bf[i].q[1] = *(const uint4*)(bb + 16 + hh * 8);
        }
#pragma unroll
        for (int i = 0; i < 8; i++)
          c[i] = __builtin_amdgcn_wmma_f32_16x16x32_bf16(
              false, a.v, false, bf[i].v, (short)0, c[i], false, false);
      }
    }
    // epilogue 2: bias + relu, weight, atomic scatter into out
#pragma unroll
    for (int i = 0; i < 8; i++) {
      int n = nchunk + (ntb + i) * 16 + ln16;
      float bias = b2[e * DDIM + n];
#pragma unroll
      for (int v = 0; v < 8; v++) {
        int m = mt * 16 + hh * 8 + v;
        float y = c[i][v] + bias;
        y = y > 0.0f ? y : 0.0f;
        y *= sWt[m];
        atomicAdd(&out[(size_t)sTok[m] * DDIM + n], y);
      }
    }
  }
}

// ---------------- launcher ----------------
extern "C" void kernel_launch(void* const* d_in, const int* in_sizes, int n_in,
                              void* d_out, int out_size, void* d_ws, size_t ws_size,
                              hipStream_t stream) {
  const float* x  = (const float*)d_in[0];
  const float* rw = (const float*)d_in[1];
  const float* w1 = (const float*)d_in[2];
  const float* b1 = (const float*)d_in[3];
  const float* w2 = (const float*)d_in[4];
  const float* b2 = (const float*)d_in[5];
  float* out = (float*)d_out;

  // workspace layout
  unsigned short* xb  = (unsigned short*)d_ws;                    // B*D bf16
  unsigned short* w1b = xb  + (size_t)BTOK * DDIM;                // E*H*D bf16
  unsigned short* w2b = w1b + (size_t)NEXP * HDIM * DDIM;         // E*D*H bf16
  int*   ecount = (int*)(w2b + (size_t)NEXP * DDIM * HDIM);       // E counters (pad 64)
  int*   etok   = ecount + 64;                                    // E*B token ids
  float* ewt    = (float*)(etok + (size_t)NEXP * BTOK);           // E*B weights

  moe_init<<<2048, 256, 0, stream>>>(out, ecount, (long)BTOK * DDIM);
  moe_cvt <<<4096, 256, 0, stream>>>(x,  xb,  (long)BTOK * DDIM);
  moe_cvt <<<2048, 256, 0, stream>>>(w1, w1b, (long)NEXP * HDIM * DDIM);
  moe_cvt <<<2048, 256, 0, stream>>>(w2, w2b, (long)NEXP * DDIM * HDIM);
  moe_router<<<BTOK / 8, 256, 0, stream>>>(x, rw, ecount, etok, ewt);

  dim3 grid(TILES_PER_E, NEXP);
  moe_expert<<<grid, 256, 0, stream>>>(xb, w1b, b1, w2b, b2, ecount, etok, ewt, out);
}
